// GCN_Net_14233521619461
// MI455X (gfx1250) — matrix operand order repaired and checked by reading, not compiled, not run on hardware
//
#include <hip/hip_runtime.h>
#include <hip/hip_bf16.h>
#include <math.h>

typedef __attribute__((ext_vector_type(16))) __bf16 v16bf;
typedef __attribute__((ext_vector_type(8)))  float  v8f;

#define HID     256
#define NCLS    40
#define BSTRIDE 40   // LDS row stride in bf16 elems (32 + 8 pad) -> 80B rows, 16B-aligned halves

// ---------------------------------------------------------------------------
// Degree / norm kernels
// ---------------------------------------------------------------------------
__global__ void deg_init_kernel(float* deg, int n) {
    int i = blockIdx.x * blockDim.x + threadIdx.x;
    if (i < n) deg[i] = 1.0f;   // self-loop contributes 1
}

__global__ void deg_acc_kernel(const int* __restrict__ dst, float* deg, int e) {
    int i = blockIdx.x * blockDim.x + threadIdx.x;
    if (i < e) atomicAdd(&deg[dst[i]], 1.0f);
}

__global__ void deg_rsqrt_kernel(float* deg, int n) {
    int i = blockIdx.x * blockDim.x + threadIdx.x;
    if (i < n) deg[i] = rsqrtf(deg[i]);
}

// ---------------------------------------------------------------------------
// GEMM: H[nrows x fout] = A[nrows x fin] * W[fin x fout]
// bf16 WMMA 16x16x32, f32 accumulate. One wave -> one 16x16 tile.
// Block = 256 threads = 8 waves laid out 4 (rows) x 2 (cols): 64x32 tile.
// Requirements: nrows % 16 == 0 (50000 ok), fin % 32 == 0, fout % 4 == 0.
// B tile staged in LDS as bf16, transposed [col][k], so fragment reads are
// two 16B ds_load_b128 per lane with no divergence.
// ---------------------------------------------------------------------------
__global__ void gemm_wmma_bf16(const float* __restrict__ A,
                               const float* __restrict__ W,
                               float* __restrict__ H,
                               int nrows, int fin, int fout) {
    __shared__ __bf16 sB[32 * BSTRIDE];

    const int lane    = threadIdx.x & 31;
    const int wave    = threadIdx.x >> 5;
    const int waveRow = wave & 3;
    const int waveCol = wave >> 2;

    const int rowBase = blockIdx.x * 64 + waveRow * 16;
    const int colBase = blockIdx.y * 32 + waveCol * 16;
    const bool waveActive = (rowBase < nrows) && (colBase < fout);

    const int half = (lane < 16) ? 0 : 1;
    const int l15  = lane & 15;

    const int m = rowBase + l15;                  // A row for this lane
    const int n = colBase + l15;                  // output column for this lane
    const bool nOk = (n < fout);

    // safe A row pointer even for inactive waves (clamped; never dereferenced
    // with bad data consumed since inactive waves skip the wmma branch)
    const int mSafe = (m < nrows) ? m : 0;
    const float* __restrict__ arow = A + (size_t)mSafe * fin;

    // B staging assignment: thread t handles 4 consecutive cols of one k row
    const int eid = threadIdx.x * 4;              // 0..1020
    const int klS = eid >> 5;                     // k row 0..31
    const int clS = eid & 31;                     // col 0..28 step 4
    const int gc  = blockIdx.y * 32 + clS;
    const int gcc = (gc < fout) ? gc : 0;         // fout%4==0 -> all-or-nothing
    const float msk = (gc < fout) ? 1.0f : 0.0f;

    v8f acc = {};

    for (int k0 = 0; k0 < fin; k0 += 32) {
        // ---- stage W[k0..k0+31][32 block cols] -> LDS bf16 transposed ----
        {
            const float4 wv = *(const float4*)(W + (size_t)(k0 + klS) * fout + gcc);
            sB[(clS + 0) * BSTRIDE + klS] = (__bf16)(wv.x * msk);
            sB[(clS + 1) * BSTRIDE + klS] = (__bf16)(wv.y * msk);
            sB[(clS + 2) * BSTRIDE + klS] = (__bf16)(wv.z * msk);
            sB[(clS + 3) * BSTRIDE + klS] = (__bf16)(wv.w * msk);
        }
        __syncthreads();

        if (waveActive) {
            // ---- A fragment: direct global float4 loads + cvt ----
            // lanes 0-15 : elems 0-7 -> K k0+0..7,  elems 8-15 -> K k0+16..23
            // lanes16-31 : elems 0-7 -> K k0+8..15, elems 8-15 -> K k0+24..31
            const int kA0 = k0 + half * 8;
            const int kA1 = kA0 + 16;
            float4 a0 = *(const float4*)(arow + kA0);
            float4 a1 = *(const float4*)(arow + kA0 + 4);
            float4 a2 = *(const float4*)(arow + kA1);
            float4 a3 = *(const float4*)(arow + kA1 + 4);

            v16bf av;
            av[0]  = (__bf16)a0.x; av[1]  = (__bf16)a0.y; av[2]  = (__bf16)a0.z; av[3]  = (__bf16)a0.w;
            av[4]  = (__bf16)a1.x; av[5]  = (__bf16)a1.y; av[6]  = (__bf16)a1.z; av[7]  = (__bf16)a1.w;
            av[8]  = (__bf16)a2.x; av[9]  = (__bf16)a2.y; av[10] = (__bf16)a2.z; av[11] = (__bf16)a2.w;
            av[12] = (__bf16)a3.x; av[13] = (__bf16)a3.y; av[14] = (__bf16)a3.z; av[15] = (__bf16)a3.w;

            // ---- B fragment: two 16B chunks 32B apart from LDS (2x ds_load_b128) ----
            // elems 0-7 -> K = k0 + half*8 + e, elems 8-15 -> K = k0 + 16 + half*8 + (e-8), column n (= waveCol*16 + l15 local)
            const int cl = waveCol * 16 + l15;
            const uint4* q = (const uint4*)&sB[cl * BSTRIDE + half * 8];
            union { uint4 u[2]; v16bf v; } bb;
            bb.u[0] = q[0];
            bb.u[1] = q[2];

            acc = __builtin_amdgcn_wmma_f32_16x16x32_bf16(
                /*neg_a=*/false, av, /*neg_b=*/false, bb.v,
                /*c_mod=*/(short)0, acc, /*reuse_a=*/false, /*reuse_b=*/false);
        }
        __syncthreads();
    }

    // C/D layout: VGPR r -> (M = rowBase + r + 8*half, N = n)
    if (waveActive && nOk) {
        const int rbase = rowBase + half * 8;
        #pragma unroll
        for (int r = 0; r < 8; ++r)
            H[(size_t)(rbase + r) * fout + n] = acc[r];
    }
}

// ---------------------------------------------------------------------------
// Aggregation: one wave per edge (grid-stride). Each lane handles 4
// consecutive features (float4 gather), atomic float scatter-add.
// Requires fout % 4 == 0.
// ---------------------------------------------------------------------------
__global__ void aggregate_kernel(const float* __restrict__ h,
                                 const int* __restrict__ src,
                                 const int* __restrict__ dst,
                                 const float* __restrict__ invs,
                                 float* __restrict__ agg,
                                 int e, int fout) {
    const int lane = threadIdx.x & 31;
    int wid  = (blockIdx.x * blockDim.x + threadIdx.x) >> 5;
    const int nw = (gridDim.x * blockDim.x) >> 5;

    for (int ei = wid; ei < e; ei += nw) {
        const int s = src[ei];
        const int d = dst[ei];
        const float w = invs[s] * invs[d];
        const float* __restrict__ hs = h + (size_t)s * fout;
        float* __restrict__ ad = agg + (size_t)d * fout;
        for (int f0 = lane * 4; f0 < fout; f0 += 128) {
            float4 v = *(const float4*)(hs + f0);
            atomicAdd(&ad[f0 + 0], v.x * w);
            atomicAdd(&ad[f0 + 1], v.y * w);
            atomicAdd(&ad[f0 + 2], v.z * w);
            atomicAdd(&ad[f0 + 3], v.w * w);
        }
    }
}

// ---------------------------------------------------------------------------
// Epilogue: out = agg + h * self_norm + bias, optional ReLU
// ---------------------------------------------------------------------------
__global__ void finish_kernel(const float* __restrict__ h,
                              const float* __restrict__ agg,
                              const float* __restrict__ invs,
                              const float* __restrict__ bias,
                              float* __restrict__ out,
                              int n, int fout, int do_relu) {
    long long i = (long long)blockIdx.x * blockDim.x + threadIdx.x;
    long long tot = (long long)n * fout;
    if (i >= tot) return;
    int node = (int)(i / fout);
    int f    = (int)(i - (long long)node * fout);
    float sn = invs[node] * invs[node];
    float v  = agg[i] + h[i] * sn + bias[f];
    out[i] = do_relu ? fmaxf(v, 0.0f) : v;
}

// ---------------------------------------------------------------------------
// Log-softmax over rows of C=40: one wave per row, shuffle reductions (wave32)
// ---------------------------------------------------------------------------
__global__ void log_softmax_kernel(const float* __restrict__ logits,
                                   float* __restrict__ out, int n, int c) {
    const int lane = threadIdx.x & 31;
    const int row  = (blockIdx.x * blockDim.x + threadIdx.x) >> 5;
    if (row >= n) return;
    const float* __restrict__ p = logits + (size_t)row * c;

    float m = -INFINITY;
    for (int j = lane; j < c; j += 32) m = fmaxf(m, p[j]);
    #pragma unroll
    for (int o = 16; o > 0; o >>= 1) m = fmaxf(m, __shfl_xor(m, o, 32));

    float s = 0.0f;
    for (int j = lane; j < c; j += 32) s += __expf(p[j] - m);
    #pragma unroll
    for (int o = 16; o > 0; o >>= 1) s += __shfl_xor(s, o, 32);

    const float lse = m + __logf(s);
    for (int j = lane; j < c; j += 32)
        out[(size_t)row * c + j] = p[j] - lse;
}

// ---------------------------------------------------------------------------
// One GCN layer
// ---------------------------------------------------------------------------
static void gcn_layer(const float* in, int fin, const float* W, const float* b,
                      int fout, const int* src, const int* dst, const float* invs,
                      float* hbuf, float* aggbuf, float* out,
                      int n, int e, bool relu, hipStream_t stream) {
    dim3 blk(256);
    dim3 grd((n + 63) / 64, (fout + 31) / 32);
    gemm_wmma_bf16<<<grd, blk, 0, stream>>>(in, W, hbuf, n, fin, fout);

    hipMemsetAsync(aggbuf, 0, (size_t)n * fout * sizeof(float), stream);

    aggregate_kernel<<<4096, 256, 0, stream>>>(hbuf, src, dst, invs, aggbuf, e, fout);

    long long tot = (long long)n * fout;
    int fblocks = (int)((tot + 255) / 256);
    finish_kernel<<<fblocks, 256, 0, stream>>>(hbuf, aggbuf, invs, b, out,
                                               n, fout, relu ? 1 : 0);
}

// ---------------------------------------------------------------------------
extern "C" void kernel_launch(void* const* d_in, const int* in_sizes, int n_in,
                              void* d_out, int out_size, void* d_ws, size_t ws_size,
                              hipStream_t stream) {
    const float* x  = (const float*)d_in[0];
    const int*   ei = (const int*)  d_in[1];
    const float* W1 = (const float*)d_in[2];
    const float* b1 = (const float*)d_in[3];
    const float* W2 = (const float*)d_in[4];
    const float* b2 = (const float*)d_in[5];
    const float* W3 = (const float*)d_in[6];
    const float* b3 = (const float*)d_in[7];
    const float* W4 = (const float*)d_in[8];
    const float* b4 = (const float*)d_in[9];

    const int F_IN = 128;
    const int n = in_sizes[0] / F_IN;       // 50000
    const int e = in_sizes[1] / 2;          // 1600000
    const int* src = ei;
    const int* dst = ei + e;

    // Workspace layout (floats)
    float* invs = (float*)d_ws;                      // n
    float* hbuf = invs + n;                          // n*HID
    float* aggb = hbuf + (size_t)n * HID;            // n*HID
    float* actA = aggb + (size_t)n * HID;            // n*HID
    float* actB = actA + (size_t)n * HID;            // n*HID

    // Degree -> inv sqrt (invs used as deg accumulator, then overwritten)
    deg_init_kernel<<<(n + 255) / 256, 256, 0, stream>>>(invs, n);
    deg_acc_kernel<<<(e + 255) / 256, 256, 0, stream>>>(dst, invs, e);
    deg_rsqrt_kernel<<<(n + 255) / 256, 256, 0, stream>>>(invs, n);

    // 4 GCN layers
    gcn_layer(x,    F_IN, W1, b1, HID,  src, dst, invs, hbuf, aggb, actA, n, e, true,  stream);
    gcn_layer(actA, HID,  W2, b2, HID,  src, dst, invs, hbuf, aggb, actB, n, e, true,  stream);
    gcn_layer(actB, HID,  W3, b3, HID,  src, dst, invs, hbuf, aggb, actA, n, e, true,  stream);
    gcn_layer(actA, HID,  W4, b4, NCLS, src, dst, invs, hbuf, aggb, actB, n, e, false, stream);

    // Log-softmax over rows of NCLS into d_out
    int rows_per_block = 256 / 32;
    int lsm_blocks = (n + rows_per_block - 1) / rows_per_block;
    log_softmax_kernel<<<lsm_blocks, 256, 0, stream>>>(actB, (float*)d_out, n, NCLS);
}
